// CausalSelfAttention_33036888441563
// MI455X (gfx1250) — matrix operand order repaired and checked by reading, not compiled, not run on hardware
//
#include <hip/hip_runtime.h>

typedef __attribute__((ext_vector_type(4)))  _Float16 v4h;
typedef __attribute__((ext_vector_type(8)))  _Float16 v8h;
typedef __attribute__((ext_vector_type(16))) _Float16 v16h;
typedef __attribute__((ext_vector_type(8)))  float    v8f;

#define WMMA_F16(a, b, c) \
  __builtin_amdgcn_wmma_f32_16x16x32_f16(false, (a), false, (b), (short)0, (c), false, false)

// Build a 16x32 A-frag / 32x16 B-frag (v16h) from two 16-byte chunks.
__device__ __forceinline__ v16h frag_ld(const _Float16* p0, const _Float16* p1) {
  v8h a = *(const v8h*)p0;
  v8h b = *(const v8h*)p1;
  v16h r;
#pragma unroll
  for (int i = 0; i < 8; ++i) { r[i] = a[i]; r[i + 8] = b[i]; }
  return r;
}

// Async 16-byte global -> LDS copy (CDNA5 path, ASYNCcnt-tracked, no VGPR data).
__device__ __forceinline__ void async_cp16(_Float16* lds, const _Float16* gptr) {
  unsigned ldsAddr = (unsigned)(size_t)lds;  // low 32 bits of generic addr = LDS offset
  asm volatile("global_load_async_to_lds_b128 %0, %1, off"
               :: "v"(ldsAddr), "v"(gptr) : "memory");
}
__device__ __forceinline__ void async_wait0() {
  asm volatile("s_wait_asynccnt 0" ::: "memory");
}

// ---------------------------------------------------------------------------
// Pre-pass 1: x (fp32) -> x_h (f16), elementwise, float4-vectorized.
// ---------------------------------------------------------------------------
__global__ __launch_bounds__(256) void convert_x_kernel(const float4* __restrict__ X,
                                                        v4h* __restrict__ Xh) {
  const size_t i = (size_t)blockIdx.x * 256 + threadIdx.x;
  float4 v = X[i];
  v4h h;
  h[0] = (_Float16)v.x; h[1] = (_Float16)v.y;
  h[2] = (_Float16)v.z; h[3] = (_Float16)v.w;
  Xh[i] = h;
}

// ---------------------------------------------------------------------------
// Pre-pass 2: W [K][N] fp32 -> Wt [N][K] f16 (32x32 LDS tile transpose).
// grid = (N/32, K/32), 256 threads.
// ---------------------------------------------------------------------------
__global__ __launch_bounds__(256) void transpose_w_kernel(const float* __restrict__ W,
                                                          _Float16* __restrict__ Wt,
                                                          int Ncols, int Krows) {
  __shared__ _Float16 T[32][33];
  const int tx = threadIdx.x & 31, ty = threadIdx.x >> 5;
  const int n0 = blockIdx.x * 32, k0 = blockIdx.y * 32;
#pragma unroll
  for (int i = 0; i < 4; ++i)
    T[ty + i * 8][tx] = (_Float16)W[(size_t)(k0 + ty + i * 8) * Ncols + n0 + tx];
  __syncthreads();
#pragma unroll
  for (int i = 0; i < 4; ++i)
    Wt[(size_t)(n0 + ty + i * 8) * Krows + k0 + tx] = T[tx][ty + i * 8];
}

// ---------------------------------------------------------------------------
// GEMM: C[8192, N] = A[8192,1024](f16) * Bt[N,1024](f16, pre-transposed) + bias
// MODE 0: N = 3072 -> writes Qh, Kh (row-major f16) and Vt ([B,H,D,T] f16)
// MODE 1: N = 1024 -> writes fp32 out + bias
// Block: 128x128 tile, 256 threads (8 waves, 4x2 wave grid, 32x64 per wave).
// All tile staging via global_load_async_to_lds_b128.
// ---------------------------------------------------------------------------
template <int MODE>
__global__ __launch_bounds__(256) void gemm_kernel(
    const _Float16* __restrict__ A, const _Float16* __restrict__ Bt,
    const float* __restrict__ bias,
    _Float16* __restrict__ Qh, _Float16* __restrict__ Kh,
    _Float16* __restrict__ Vt, float* __restrict__ Out) {
  __shared__ _Float16 As[128][40];  // [row][k], pad 32->40 halfs (80B row stride)
  __shared__ _Float16 Bs[128][40];  // [n][k]

  const int tid = threadIdx.x;
  const int wave = tid >> 5, lane = tid & 31;
  const int lm = lane & 15, lh = lane >> 4;
  const int wm = wave & 3, wn = wave >> 2;
  const int blockM = blockIdx.y * 128;
  const int blockN = blockIdx.x * 128;

  const int ldRow = tid >> 2;          // 0..63
  const int ldCol = (tid & 3) * 8;     // 0,8,16,24

  v8f zero = {};
  v8f acc[2][4];
#pragma unroll
  for (int i = 0; i < 2; ++i)
#pragma unroll
    for (int j = 0; j < 4; ++j) acc[i][j] = zero;

  for (int k0 = 0; k0 < 1024; k0 += 32) {
#pragma unroll
    for (int c = 0; c < 2; ++c) {
      const int row = ldRow + c * 64;
      async_cp16(&As[row][ldCol], A + (size_t)(blockM + row) * 1024 + k0 + ldCol);
      async_cp16(&Bs[row][ldCol], Bt + (size_t)(blockN + row) * 1024 + k0 + ldCol);
    }
    async_wait0();
    __syncthreads();

    v16h aF[2], bF[4];
#pragma unroll
    for (int i = 0; i < 2; ++i) {
      const _Float16* r0 = &As[wm * 32 + i * 16 + lm][lh * 8];
      const _Float16* r1 = &As[wm * 32 + i * 16 + lm][16 + lh * 8];
      aF[i] = frag_ld(r0, r1);
    }
#pragma unroll
    for (int j = 0; j < 4; ++j) {
      const _Float16* r0 = &Bs[wn * 64 + j * 16 + lm][lh * 8];
      bF[j] = frag_ld(r0, r0 + 16);
    }
#pragma unroll
    for (int i = 0; i < 2; ++i)
#pragma unroll
      for (int j = 0; j < 4; ++j) acc[i][j] = WMMA_F16(aF[i], bF[j], acc[i][j]);
    __syncthreads();
  }

#pragma unroll
  for (int i = 0; i < 2; ++i) {
    const int rowBase = blockM + wm * 32 + i * 16 + lh * 8;
#pragma unroll
    for (int j = 0; j < 4; ++j) {
      const int ncol = blockN + wn * 64 + j * 16 + lm;
      const float bv = bias[ncol];
#pragma unroll
      for (int r = 0; r < 8; ++r) {
        const int row = rowBase + r;
        const float val = acc[i][j][r] + bv;
        if (MODE == 0) {
          const _Float16 hv = (_Float16)val;
          if (ncol < 1024) {
            Qh[(size_t)row * 1024 + ncol] = hv;
          } else if (ncol < 2048) {
            Kh[(size_t)row * 1024 + (ncol - 1024)] = hv;
          } else {
            const int dd = ncol - 2048;
            const int bI = row >> 11, t = row & 2047;
            Vt[((size_t)(bI * 16 + (dd >> 6)) * 64 + (dd & 63)) * 2048 + t] = hv;
          }
        } else {
          Out[(size_t)row * 1024 + ncol] = val;
        }
      }
    }
  }
}

// ---------------------------------------------------------------------------
// Flash attention: grid (T/128, B*H), 256 threads. Each wave owns 16 query
// rows; online softmax in fp32; QK^T and PV on v_wmma_f32_16x16x32_f16.
// ---------------------------------------------------------------------------
__global__ __launch_bounds__(256) void attn_kernel(
    const _Float16* __restrict__ Qh, const _Float16* __restrict__ Kh,
    const _Float16* __restrict__ Vt, _Float16* __restrict__ Ah) {
  __shared__ _Float16 Ps[8][16][40];  // per-wave P transpose buffer

  const int tid = threadIdx.x;
  const int wave = tid >> 5, lane = tid & 31;
  const int lm = lane & 15, lh = lane >> 4;
  const int bb = blockIdx.y >> 4, hd = blockIdx.y & 15;
  const int qs = blockIdx.x * 128 + wave * 16;

  const _Float16* qp = Qh + (size_t)(bb * 2048 + qs + lm) * 1024 + hd * 64;
  v16h qf[2];
#pragma unroll
  for (int g = 0; g < 2; ++g)
    qf[g] = frag_ld(qp + g * 32 + lh * 8, qp + g * 32 + 16 + lh * 8);

  float mr[8], lr[8];
  v8f zero = {};
  v8f o[4];
#pragma unroll
  for (int r = 0; r < 8; ++r) { mr[r] = -3.0e38f; lr[r] = 0.0f; }
#pragma unroll
  for (int g = 0; g < 4; ++g) o[g] = zero;

  const float scale = 0.125f;  // 1/sqrt(64)
  const int kend = qs + 16;    // causal: kb stays <= qs (mult of 32), every row keeps >=1 key
  for (int kb = 0; kb < kend; kb += 32) {
    v8f s0 = zero, s1 = zero;
#pragma unroll
    for (int gg = 0; gg < 2; ++gg) {
      const _Float16* kp0 =
          Kh + (size_t)(bb * 2048 + kb + lm) * 1024 + hd * 64 + gg * 32 + lh * 8;
      v16h kf0 = frag_ld(kp0, kp0 + 16);
      s0 = WMMA_F16(qf[gg], kf0, s0);
      const _Float16* kp1 =
          Kh + (size_t)(bb * 2048 + kb + 16 + lm) * 1024 + hd * 64 + gg * 32 + lh * 8;
      v16h kf1 = frag_ld(kp1, kp1 + 16);
      s1 = WMMA_F16(qf[gg], kf1, s1);
    }
    const int key0 = kb + lm, key1 = kb + 16 + lm;
#pragma unroll
    for (int r = 0; r < 8; ++r) {
      const int q = qs + r + 8 * lh;
      float sv0 = s0[r] * scale; if (key0 > q) sv0 = -3.0e38f;
      float sv1 = s1[r] * scale; if (key1 > q) sv1 = -3.0e38f;
      float mx = fmaxf(sv0, sv1);
#pragma unroll
      for (int off = 1; off < 16; off <<= 1) mx = fmaxf(mx, __shfl_xor(mx, off, 32));
      const float mn = fmaxf(mr[r], mx);
      const float alpha = __expf(mr[r] - mn);
      const float p0 = __expf(sv0 - mn);
      const float p1 = __expf(sv1 - mn);
      float rs = p0 + p1;
#pragma unroll
      for (int off = 1; off < 16; off <<= 1) rs += __shfl_xor(rs, off, 32);
      lr[r] = lr[r] * alpha + rs;
      mr[r] = mn;
#pragma unroll
      for (int g = 0; g < 4; ++g) o[g][r] *= alpha;
      Ps[wave][r + 8 * lh][lm] = (_Float16)p0;
      Ps[wave][r + 8 * lh][16 + lm] = (_Float16)p1;
    }
    asm volatile("s_wait_dscnt 0" ::: "memory");  // wave-private LDS: stores -> loads
    v16h pf = frag_ld(&Ps[wave][lm][lh * 8], &Ps[wave][lm][16 + lh * 8]);
#pragma unroll
    for (int g = 0; g < 4; ++g) {
      const _Float16* vp =
          Vt + ((size_t)(bb * 16 + hd) * 64 + g * 16 + lm) * 2048 + kb + lh * 8;
      v16h vf = frag_ld(vp, vp + 16);
      o[g] = WMMA_F16(pf, vf, o[g]);
    }
  }

#pragma unroll
  for (int r = 0; r < 8; ++r) {
    const float inv = 1.0f / lr[r];
    const size_t rowOff = (size_t)(bb * 2048 + qs + r + 8 * lh) * 1024 + hd * 64;
#pragma unroll
    for (int g = 0; g < 4; ++g)
      Ah[rowOff + g * 16 + lm] = (_Float16)(o[g][r] * inv);
  }
}

extern "C" void kernel_launch(void* const* d_in, const int* in_sizes, int n_in,
                              void* d_out, int out_size, void* d_ws, size_t ws_size,
                              hipStream_t stream) {
  (void)in_sizes; (void)n_in; (void)out_size; (void)ws_size;
  const float* x     = (const float*)d_in[0];
  const float* w_qkv = (const float*)d_in[1];
  const float* b_qkv = (const float*)d_in[2];
  const float* w_out = (const float*)d_in[3];
  const float* b_out = (const float*)d_in[4];
  float* out = (float*)d_out;

  char* ws = (char*)d_ws;
  const size_t MB = (size_t)1024 * 1024;
  _Float16* Xh    = (_Float16*)(ws + 0 * MB);   // 16 MB; dead after QKV GEMM
  _Float16* Ah    = (_Float16*)(ws + 0 * MB);   // reuses Xh segment (stream-ordered)
  _Float16* Qh    = (_Float16*)(ws + 16 * MB);  // 16 MB
  _Float16* Kh    = (_Float16*)(ws + 32 * MB);  // 16 MB
  _Float16* Vt    = (_Float16*)(ws + 48 * MB);  // 16 MB, [B,H,D,T]
  _Float16* WqkvT = (_Float16*)(ws + 64 * MB);  // 6 MB, [3072][1024]
  _Float16* WoutT = (_Float16*)(ws + 70 * MB);  // 2 MB, [1024][1024]

  // Pre-passes: cast x to f16; transpose+cast weights (all bandwidth-trivial).
  convert_x_kernel<<<8192, 256, 0, stream>>>((const float4*)x, (v4h*)Xh);
  transpose_w_kernel<<<dim3(3072 / 32, 1024 / 32), 256, 0, stream>>>(w_qkv, WqkvT, 3072, 1024);
  transpose_w_kernel<<<dim3(1024 / 32, 1024 / 32), 256, 0, stream>>>(w_out, WoutT, 1024, 1024);

  // 1) QKV projection: [8192,1024] x [1024,3072] + b_qkv
  gemm_kernel<0><<<dim3(3072 / 128, 8192 / 128), 256, 0, stream>>>(
      Xh, WqkvT, b_qkv, Qh, Kh, Vt, nullptr);

  // 2) Causal flash attention
  attn_kernel<<<dim3(2048 / 128, 4 * 16), 256, 0, stream>>>(Qh, Kh, Vt, Ah);

  // 3) Output projection: [8192,1024] x [1024,1024] + b_out -> fp32
  gemm_kernel<1><<<dim3(1024 / 128, 8192 / 128), 256, 0, stream>>>(
      Ah, WoutT, b_out, nullptr, nullptr, nullptr, out);
}
